// GroupedQueryAttention_64476049047586
// MI455X (gfx1250) — matrix operand order repaired and checked by reading, not compiled, not run on hardware
//
#include <hip/hip_runtime.h>
#include <hip/hip_bf16.h>
#include <stdint.h>

#define Bv 2
#define Tv 2048
#define Dv 4096
#define Hv 32
#define KHv 8
#define HDv 128
#define WINDOWv 1024
#define NGLOBALv 32

typedef __attribute__((ext_vector_type(16))) __bf16 v16bf;
typedef __attribute__((ext_vector_type(8)))  __bf16 v8bf;
typedef __attribute__((ext_vector_type(8)))  float  v8f;
typedef __attribute__((ext_vector_type(4)))  float  v4f;
typedef __attribute__((ext_vector_type(4)))  int    v4i;

// address-space-qualified pointer types for the async-to-LDS builtin
typedef v4i __attribute__((address_space(1)))* gas_v4i_p;  // global
typedef v4i __attribute__((address_space(3)))* las_v4i_p;  // LDS

union FragBF {
  v16bf v;
  v8bf  h[2];
};

union PairBF {
  __bf16  b[2];
  uint32_t u;
};

// hardware f32 -> bf16 (lowers to v_cvt_pk_bf16_f32 on gfx1250)
static __device__ inline __bf16 f2bf(float f) { return (__bf16)f; }

static __device__ inline v8f wmma_bf16(v16bf a, v16bf b, v8f c) {
  // D(f32 16x16) = A(bf16 16x32) * B(bf16 32x16) + C
  return __builtin_amdgcn_wmma_f32_16x16x32_bf16(
      /*neg_a=*/false, a, /*neg_b=*/false, b,
      /*c_mod=*/(short)0, c, /*reuse_a=*/false, /*reuse_b=*/false);
}

// reductions across a 16-lane group (rows of a WMMA C tile span 16 lanes)
static __device__ inline float red16max(float v) {
  v = fmaxf(v, __shfl_xor(v, 1, 32));
  v = fmaxf(v, __shfl_xor(v, 2, 32));
  v = fmaxf(v, __shfl_xor(v, 4, 32));
  v = fmaxf(v, __shfl_xor(v, 8, 32));
  return v;
}
static __device__ inline float red16sum(float v) {
  v += __shfl_xor(v, 1, 32);
  v += __shfl_xor(v, 2, 32);
  v += __shfl_xor(v, 4, 32);
  v += __shfl_xor(v, 8, 32);
  return v;
}

// Async global->LDS copy (CDNA5 GLOBAL_LOAD_ASYNC_TO_LDS_B128, ASYNCcnt) with
// synchronous fallback if the builtin is not declared by this toolchain.
#if defined(__has_builtin)
#if __has_builtin(__builtin_amdgcn_global_load_async_to_lds_b128)
#define HAVE_ASYNC_LDS 1
#endif
#endif

static __device__ inline void copy16_g2l(const __bf16* g, __bf16* l) {
#ifdef HAVE_ASYNC_LDS
  __builtin_amdgcn_global_load_async_to_lds_b128(
      (gas_v4i_p)g, (las_v4i_p)l, 0, 0);
#else
  *reinterpret_cast<v8bf*>(l) = *reinterpret_cast<const v8bf*>(g);
#endif
}

static __device__ inline void wait_async_lds() {
#ifdef HAVE_ASYNC_LDS
#if __has_builtin(__builtin_amdgcn_s_wait_asynccnt)
  __builtin_amdgcn_s_wait_asynccnt(0);
#else
  asm volatile("s_wait_asynccnt 0" ::: "memory");
#endif
#endif
}

// ---------------------------------------------------------------------------
// Generic C = A * B, row-major fp32 in/out, bf16 WMMA compute.
// Block: 256 threads (8 waves). Block tile 256(M) x 64(N). K step 32.
// Each wave owns a 32x64 output tile (2 M-subtiles x 4 N-chunks = 8 WMMAs per
// K-step against 4 shared B-fragments -> 1 ds_load_b128 per WMMA).
// Double-buffered B tile in LDS: one barrier per K-step.
// ---------------------------------------------------------------------------
__global__ __launch_bounds__(256) void gemm_bf16_kernel(
    const float* __restrict__ A, const float* __restrict__ Bm,
    float* __restrict__ C, int M, int N, int K) {
  // B tile stored transposed: ldsB[buf][n][k], padded row (40 bf16 = 80B)
  __shared__ __align__(16) __bf16 ldsB[2][64][40];

  const int tid  = threadIdx.x;
  const int w    = tid >> 5;
  const int lane = tid & 31;
  const int ln   = lane & 15;   // row (A) / col (B,C) within 16
  const int kh   = lane >> 4;   // which K half the lane holds

  const int mbase = blockIdx.y * 256 + w * 32;
  const int nbase = blockIdx.x * 64;
  const float* arow0 = A + (size_t)(mbase + ln) * K;
  const float* arow1 = A + (size_t)(mbase + 16 + ln) * K;

  // loader role: thread owns K-pair (2*kp, 2*kp+1) x 4 consecutive N
  const int kp = tid >> 4;  // 0..15
  const int n4 = tid & 15;  // 0..15

  v8f acc[2][4] = {};

  // cooperative B tile fill: packed bf16 pairs along K, one b32 store each
  auto fill = [&](int kb, int buf) {
    const float* s0 = Bm + (size_t)(kb + 2 * kp) * N + nbase + n4 * 4;
    const float* s1 = s0 + N;
    v4f r0 = *reinterpret_cast<const v4f*>(s0);
    v4f r1 = *reinterpret_cast<const v4f*>(s1);
    float e0[4] = {r0.x, r0.y, r0.z, r0.w};
    float e1[4] = {r1.x, r1.y, r1.z, r1.w};
    for (int i = 0; i < 4; ++i) {
      PairBF p; p.b[0] = f2bf(e0[i]); p.b[1] = f2bf(e1[i]);
      *reinterpret_cast<uint32_t*>(&ldsB[buf][n4 * 4 + i][2 * kp]) = p.u;
    }
    if (kb + 64 < K) __builtin_prefetch(s0 + (size_t)64 * N, 0, 1);
  };

  // A fragment loader: lane row, K = {kh*8..+8} and {16+kh*8..+8} of step
  auto loadA = [&](const float* row, int kb, FragBF& fa) {
    const float* p0 = row + kb + kh * 8;
    const float* p1 = row + kb + 16 + kh * 8;
    v4f a0 = *reinterpret_cast<const v4f*>(p0);
    v4f a1 = *reinterpret_cast<const v4f*>(p0 + 4);
    v4f a2 = *reinterpret_cast<const v4f*>(p1);
    v4f a3 = *reinterpret_cast<const v4f*>(p1 + 4);
    float av[16] = {a0.x, a0.y, a0.z, a0.w, a1.x, a1.y, a1.z, a1.w,
                    a2.x, a2.y, a2.z, a2.w, a3.x, a3.y, a3.z, a3.w};
    for (int j = 0; j < 16; ++j) fa.v[j] = f2bf(av[j]);
  };

  fill(0, 0);
  const int steps = K >> 5;
  for (int i = 0; i < steps; ++i) {
    __syncthreads();  // buf[i&1] fully written; prior reads of buf[(i+1)&1] done
    if (i + 1 < steps) fill((i + 1) * 32, (i + 1) & 1);

    const int kb = i * 32, buf = i & 1;

    FragBF fa[2];
    loadA(arow0, kb, fa[0]);
    loadA(arow1, kb, fa[1]);

    FragBF fb[4];
    for (int c = 0; c < 4; ++c) {
      const __bf16* bp = &ldsB[buf][c * 16 + ln][kh * 8];
      fb[c].h[0] = *reinterpret_cast<const v8bf*>(bp);
      fb[c].h[1] = *reinterpret_cast<const v8bf*>(bp + 16);
    }
    for (int m = 0; m < 2; ++m)
      for (int c = 0; c < 4; ++c)
        acc[m][c] = wmma_bf16(fa[m].v, fb[c].v, acc[m][c]);
  }

  for (int m = 0; m < 2; ++m) {
    float* crow = C + (size_t)(mbase + m * 16 + 8 * kh) * N + nbase + ln;
    for (int r = 0; r < 8; ++r)
      for (int c = 0; c < 4; ++c)
        crow[(size_t)r * N + c * 16] = acc[m][c][r];
  }
}

// ---------------------------------------------------------------------------
// RoPE + pack Q/K/V fp32 [token, head*HD] -> bf16 [B, heads, T, HD]
// ---------------------------------------------------------------------------
__global__ __launch_bounds__(128) void rope_pack_kernel(
    const float* __restrict__ qf, const float* __restrict__ kf,
    const float* __restrict__ vf, const float* __restrict__ cosp,
    const float* __restrict__ sinp, __bf16* __restrict__ qb,
    __bf16* __restrict__ kb, __bf16* __restrict__ vb) {
  const int token = blockIdx.x;      // b*T + t
  const int b = token / Tv, t = token % Tv;
  const int d = threadIdx.x;         // 0..127
  const float cv = cosp[t * HDv + d];
  const float sv = sinp[t * HDv + d];
  for (int h = 0; h < Hv; ++h) {
    const size_t base = (size_t)token * (Hv * HDv) + h * HDv;
    float x0 = qf[base + d];
    float xr = (d < 64) ? -qf[base + d + 64] : qf[base + d - 64];
    qb[((size_t)(b * Hv + h) * Tv + t) * HDv + d] = f2bf(x0 * cv + xr * sv);
  }
  for (int h = 0; h < KHv; ++h) {
    const size_t base = (size_t)token * (KHv * HDv) + h * HDv;
    float x0 = kf[base + d];
    float xr = (d < 64) ? -kf[base + d + 64] : kf[base + d - 64];
    const size_t ob = ((size_t)(b * KHv + h) * Tv + t) * HDv + d;
    kb[ob] = f2bf(x0 * cv + xr * sv);
    vb[ob] = f2bf(vf[base + d]);
  }
}

// ---------------------------------------------------------------------------
// Flash-style attention. Block = 64 queries of one (b, h); 4 waves x 16 q.
// Key blocks of 32: sink [0,32) + sliding window. Online softmax.
// K tile copied with GLOBAL_LOAD_ASYNC_TO_LDS_B128 (ASYNCcnt-tracked).
// ---------------------------------------------------------------------------
__global__ __launch_bounds__(128) void attn_kernel(
    const __bf16* __restrict__ Qb, const __bf16* __restrict__ Kb,
    const __bf16* __restrict__ Vb, float* __restrict__ O) {
  __shared__ __align__(16) __bf16 ldsK[32][136];   // [key][hd], padded
  __shared__ __align__(16) __bf16 ldsVt[128][40];  // [hd][key], padded
  __shared__ __align__(16) __bf16 ldsP[4][16][40]; // per-wave P (16q x 32k)

  const int bh = blockIdx.x;
  const int b = bh / Hv, h = bh % Hv;
  const int kvh = h / (Hv / KHv);     // GQA: 4 query heads per KV head
  const int q0 = blockIdx.y * 64;
  const int tid = threadIdx.x;
  const int w = tid >> 5, lane = tid & 31;
  const int ln = lane & 15, kh = lane >> 4;
  const int qw = q0 + w * 16;

  // Q fragments (16 queries x 128 hd = 4 WMMA K-steps), kept in registers
  FragBF qfr[4];
  {
    const __bf16* qrow = Qb + ((size_t)(b * Hv + h) * Tv + qw + ln) * HDv;
    for (int s = 0; s < 4; ++s) {
      qfr[s].h[0] = *reinterpret_cast<const v8bf*>(qrow + s * 32 + kh * 16);
      qfr[s].h[1] = *reinterpret_cast<const v8bf*>(qrow + s * 32 + kh * 16 + 8);
    }
  }

  float rowM[8], rowL[8];
  v8f oacc[8] = {};
  for (int r = 0; r < 8; ++r) { rowM[r] = -__builtin_inff(); rowL[r] = 0.f; }

  int jlo = q0 - (WINDOWv - 1);
  if (jlo < NGLOBALv) jlo = NGLOBALv;
  const int kb_lo = jlo & ~31;
  const int kb_hi = q0 + 63;
  const float scale = 0.08838834764831845f; // 1/sqrt(128)

  const __bf16* kgbase = Kb + (size_t)(b * KHv + kvh) * Tv * HDv;
  const __bf16* vgbase = Vb + (size_t)(b * KHv + kvh) * Tv * HDv;

  // per-thread roles in the cooperative loaders
  const int krow = tid >> 2, kseg = tid & 3;  // K: 32 rows x 4 segs of 32 hd
  const int vkp = tid >> 3, vseg = tid & 7;   // V: 16 key-pairs x 8 segs of 16 hd

  for (int blk = -1;; ++blk) {
    const int kb = (blk < 0) ? 0 : kb_lo + blk * 32;
    if (blk >= 0 && kb > kb_hi) break;

    __syncthreads();
    {
      // K tile: async DMA straight into the [key][hd] layout
      const __bf16* kg = kgbase + (size_t)(kb + krow) * HDv + kseg * 32;
      for (int c = 0; c < 4; ++c)
        copy16_g2l(kg + c * 8, &ldsK[krow][kseg * 32 + c * 8]);

      // V tile: transpose into [hd][key], two keys packed per 32-bit store
      const __bf16* vg0 = vgbase + (size_t)(kb + 2 * vkp) * HDv + vseg * 16;
      const __bf16* vg1 = vg0 + HDv;
      v8bf va0 = *reinterpret_cast<const v8bf*>(vg0);
      v8bf va1 = *reinterpret_cast<const v8bf*>(vg0 + 8);
      v8bf vb0 = *reinterpret_cast<const v8bf*>(vg1);
      v8bf vb1 = *reinterpret_cast<const v8bf*>(vg1 + 8);
      for (int i = 0; i < 8; ++i) {
        PairBF p0; p0.b[0] = va0[i]; p0.b[1] = vb0[i];
        PairBF p1; p1.b[0] = va1[i]; p1.b[1] = vb1[i];
        *reinterpret_cast<uint32_t*>(&ldsVt[vseg * 16 + i][2 * vkp]) = p0.u;
        *reinterpret_cast<uint32_t*>(&ldsVt[vseg * 16 + 8 + i][2 * vkp]) = p1.u;
      }

      // prefetch next key block while this one is consumed
      const int nkb = (blk < 0) ? kb_lo : kb + 32;
      if (nkb <= kb_hi) {
        __builtin_prefetch(kgbase + (size_t)(nkb + krow) * HDv + kseg * 32, 0, 1);
        __builtin_prefetch(vgbase + (size_t)(nkb + krow) * HDv + kseg * 32, 0, 1);
      }
    }
    wait_async_lds();
    __syncthreads();

    // S(16q x 32k) = Q * K^T: load all 8 K fragments, then burst 8 WMMAs
    FragBF kf0[4], kf1[4];
    for (int s = 0; s < 4; ++s) {
      const __bf16* kp0 = &ldsK[ln][s * 32 + kh * 16];
      const __bf16* kp1 = &ldsK[16 + ln][s * 32 + kh * 16];
      kf0[s].h[0] = *reinterpret_cast<const v8bf*>(kp0);
      kf0[s].h[1] = *reinterpret_cast<const v8bf*>(kp0 + 8);
      kf1[s].h[0] = *reinterpret_cast<const v8bf*>(kp1);
      kf1[s].h[1] = *reinterpret_cast<const v8bf*>(kp1 + 8);
    }
    v8f s0 = {}, s1 = {};
    for (int s = 0; s < 4; ++s) {
      s0 = wmma_bf16(qfr[s].v, kf0[s].v, s0);
      s1 = wmma_bf16(qfr[s].v, kf1[s].v, s1);
    }

    // mask + online softmax (C layout: col = ln, row = r + 8*kh)
    const int j0 = kb + ln, j1 = kb + 16 + ln;
    for (int r = 0; r < 8; ++r) {
      const int qq = qw + r + 8 * kh;
      float a  = s0[r] * scale;
      float bb = s1[r] * scale;
      const bool ok0 = (j0 <= qq) && ((j0 > qq - WINDOWv) || (j0 < NGLOBALv));
      const bool ok1 = (j1 <= qq) && ((j1 > qq - WINDOWv) || (j1 < NGLOBALv));
      a  = ok0 ? a  : -3.0e38f;
      bb = ok1 ? bb : -3.0e38f;
      const float mt   = red16max(fmaxf(a, bb));
      const float mnew = fmaxf(rowM[r], mt);
      const float p0 = __expf(a - mnew);
      const float p1 = __expf(bb - mnew);
      const float st = red16sum(p0 + p1);
      const float alpha = __expf(rowM[r] - mnew);
      rowL[r] = rowL[r] * alpha + st;
      rowM[r] = mnew;
      for (int c = 0; c < 8; ++c) oacc[c][r] *= alpha;
      ldsP[w][r + 8 * kh][ln]      = f2bf(p0);
      ldsP[w][r + 8 * kh][16 + ln] = f2bf(p1);
    }

    // O += P * V: load P + all 8 V fragments, then burst 8 WMMAs
    FragBF pf;
    {
      const __bf16* pp = &ldsP[w][ln][0];
      pf.h[0] = *reinterpret_cast<const v8bf*>(pp + kh * 16);
      pf.h[1] = *reinterpret_cast<const v8bf*>(pp + kh * 16 + 8);
    }
    FragBF vfr[8];
    for (int c = 0; c < 8; ++c) {
      const __bf16* vp = &ldsVt[c * 16 + ln][kh * 16];
      vfr[c].h[0] = *reinterpret_cast<const v8bf*>(vp);
      vfr[c].h[1] = *reinterpret_cast<const v8bf*>(vp + 8);
    }
    for (int c = 0; c < 8; ++c) oacc[c] = wmma_bf16(pf.v, vfr[c].v, oacc[c]);
  }

  float* orow = O + (size_t)(b * Tv + qw + 8 * kh) * Dv + h * HDv + ln;
  for (int r = 0; r < 8; ++r)
    for (int c = 0; c < 8; ++c)
      orow[(size_t)r * Dv + c * 16] = oacc[c][r] / rowL[r];
}

// ---------------------------------------------------------------------------
extern "C" void kernel_launch(void* const* d_in, const int* in_sizes, int n_in,
                              void* d_out, int out_size, void* d_ws,
                              size_t ws_size, hipStream_t stream) {
  (void)in_sizes; (void)n_in; (void)out_size; (void)ws_size;
  const float* x    = (const float*)d_in[0];
  const float* Wq   = (const float*)d_in[1];
  const float* Wk   = (const float*)d_in[2];
  const float* Wv   = (const float*)d_in[3];
  const float* Wo   = (const float*)d_in[4];
  const float* cosp = (const float*)d_in[5];
  const float* sinp = (const float*)d_in[6];
  // d_in[7] (bool mask) ignored: mask is computed analytically in-kernel.
  float* out = (float*)d_out;

  char* w = (char*)d_ws;
  const size_t MiB = (size_t)1 << 20;
  float*  qf  = (float*)(w + 0);          // 64 MiB  [4096 x 4096] f32
  float*  kf  = (float*)(w + 64  * MiB);  // 16 MiB  [4096 x 1024] f32
  float*  vf  = (float*)(w + 80  * MiB);  // 16 MiB
  __bf16* qbw = (__bf16*)(w + 96 * MiB);  // 32 MiB  [B,H,T,HD]  bf16
  __bf16* kbw = (__bf16*)(w + 128 * MiB); //  8 MiB  [B,KH,T,HD] bf16
  __bf16* vbw = (__bf16*)(w + 136 * MiB); //  8 MiB
  float*  aO  = qf;  // reuse: qf is dead after rope_pack

  const int M = Bv * Tv;  // 4096 tokens

  gemm_bf16_kernel<<<dim3(Dv / 64, M / 256), 256, 0, stream>>>(
      x, Wq, qf, M, Dv, Dv);
  gemm_bf16_kernel<<<dim3((KHv * HDv) / 64, M / 256), 256, 0, stream>>>(
      x, Wk, kf, M, KHv * HDv, Dv);
  gemm_bf16_kernel<<<dim3((KHv * HDv) / 64, M / 256), 256, 0, stream>>>(
      x, Wv, vf, M, KHv * HDv, Dv);
  rope_pack_kernel<<<dim3(M), 128, 0, stream>>>(
      qf, kf, vf, cosp, sinp, qbw, kbw, vbw);
  attn_kernel<<<dim3(Bv * Hv, Tv / 64), 128, 0, stream>>>(
      qbw, kbw, vbw, aO);
  gemm_bf16_kernel<<<dim3(Dv / 64, M / 256), 256, 0, stream>>>(
      aO, Wo, out, M, Dv, Dv);
}
